// ExpertTransformerBlock_49469433315516
// MI455X (gfx1250) — compile-verified, matrix operand order checked by reading, not run on hardware
//
#include <hip/hip_runtime.h>
#include <stdint.h>

#define DEV __device__ __forceinline__

#define Bb   128
#define Ss   64
#define Dd   512
#define Hh   8
#define DHh  64
#define EAx  8
#define EFx  8
#define DFF  2048
#define TOKS (Bb*Ss)          // 8192
#define EPSV 1e-5f

typedef __attribute__((ext_vector_type(16))) __bf16 v16bf;
typedef __attribute__((ext_vector_type(8)))  float  v8f;

// ---------- bf16 helpers: native f32->bf16 convert (RNE in HW) ----------
DEV unsigned short f2bf_u(float f) {
  __bf16 b = (__bf16)f;
  unsigned short r; __builtin_memcpy(&r, &b, 2);
  return r;
}
// pack two floats into one dword of bf16 (fuses to v_cvt_pk_bf16_f32)
DEV unsigned pack2(float x, float y) {
  return (unsigned)f2bf_u(x) | ((unsigned)f2bf_u(y) << 16);
}

// ---------- WMMA fragment loaders ----------
// A-fragment layout (16x32 bf16, ISA 7.12.2): lane L: row = row0 + (L&15),
// chunk0 = K[kbase + (L>>4)*8 .. +7], chunk1 = K[kbase+16+(L>>4)*8 .. +7].
// B-fragment mirrors A with lane = output column, reading rows of W ([N,K]).
union FragCvt { v16bf v; struct { uint4 c0; uint4 c1; } q; };

DEV v16bf frag_from_chunks(uint4 a, uint4 b) { FragCvt u; u.q.c0 = a; u.q.c1 = b; return u.v; }

DEV v16bf load_frag_rowp(const unsigned short* rowp, int kbase, int hi) {
  const unsigned short* q = rowp + kbase + hi * 8;
  return frag_from_chunks(*(const uint4*)q, *(const uint4*)(q + 16));
}
DEV v16bf load_frag_bf(const unsigned short* p, int ld, int row0, int kbase, int lo, int hi) {
  return load_frag_rowp(p + (size_t)(row0 + lo) * ld, kbase, hi);
}
DEV v8f wmma_bf(v16bf a, v16bf b, v8f c) {
  return __builtin_amdgcn_wmma_f32_16x16x32_bf16(false, a, false, b, (short)0, c, false, false);
}
DEV v8f zero8() { v8f z = {0.f,0.f,0.f,0.f,0.f,0.f,0.f,0.f}; return z; }

// ======================================================================
// 0) weight conversion: fp32 -> bf16 (done once per launch, amortized
//    over ~32 reuses of every expert weight; halves hot-loop bytes)
// ======================================================================
__global__ __launch_bounds__(256) void k_cvt(
    const float* __restrict__ s, unsigned* __restrict__ d, size_t npairs) {
  size_t stride = (size_t)gridDim.x * blockDim.x;
  for (size_t i = (size_t)blockIdx.x * blockDim.x + threadIdx.x;
       i < npairs; i += stride) {
    float2 f = *(const float2*)(s + i * 2);
    d[i] = pack2(f.x, f.y);
  }
}

// ======================================================================
// 1) prep: x -> bf16, y1 = x (residual base), zero small accumulators
// ======================================================================
__global__ __launch_bounds__(256) void k_prep(
    const float* __restrict__ x, unsigned short* __restrict__ xbf,
    float* __restrict__ y1, float* out_pa, float* out_pf,
    float* sumA, float* sqA, float* sumF, float* sqF, int* cnt) {
  size_t stride = (size_t)gridDim.x * blockDim.x;
  for (size_t i = (size_t)blockIdx.x * blockDim.x + threadIdx.x;
       i < (size_t)TOKS * Dd; i += stride) {
    float v = x[i];
    y1[i]  = v;
    xbf[i] = f2bf_u(v);
  }
  if (blockIdx.x == 0) {
    int t = threadIdx.x;
    if (t < Dd) { sumA[t] = 0.f; sqA[t] = 0.f; sumF[t] = 0.f; sqF[t] = 0.f; out_pf[t] = 0.f; }
    if (t < EAx) { out_pa[t] = 0.f; cnt[t] = 0; }
  }
}

// ======================================================================
// 2) attention router: per-sample mean -> softmax -> top2
// ======================================================================
__global__ __launch_bounds__(256) void k_router_attn(
    const float* __restrict__ x, const float* __restrict__ Wra,
    const float* __restrict__ bra, int* aidx, float* aw, float* out_pa) {
  __shared__ float xm[Dd];
  __shared__ float lg[EAx];
  int b = blockIdx.x;
  for (int d = threadIdx.x; d < Dd; d += blockDim.x) {
    float s = 0.f;
    const float* p = x + ((size_t)b * Ss) * Dd + d;
    for (int ss = 0; ss < Ss; ++ss) s += p[(size_t)ss * Dd];
    xm[d] = s * (1.0f / Ss);
  }
  __syncthreads();
  if (threadIdx.x < EAx) {
    int e = threadIdx.x;
    float s = bra[e];
    const float* wr = Wra + e * Dd;
    for (int d = 0; d < Dd; ++d) s += xm[d] * wr[d];
    lg[e] = s;
  }
  __syncthreads();
  if (threadIdx.x == 0) {
    float mx = lg[0];
    for (int e = 1; e < EAx; ++e) mx = fmaxf(mx, lg[e]);
    float p[EAx]; float ss = 0.f;
    for (int e = 0; e < EAx; ++e) { p[e] = expf(lg[e] - mx); ss += p[e]; }
    float inv = 1.0f / ss;
    for (int e = 0; e < EAx; ++e) p[e] *= inv;
    int i0 = 0;
    for (int e = 1; e < EAx; ++e) if (p[e] > p[i0]) i0 = e;
    int i1 = (i0 == 0) ? 1 : 0;
    for (int e = 0; e < EAx; ++e) if (e != i0 && p[e] > p[i1]) i1 = e;
    for (int e = 0; e < EAx; ++e) atomicAdd(&out_pa[e], p[e] * (1.0f / Bb));
    float vs = p[i0] + p[i1];
    aidx[b * 2 + 0] = i0; aidx[b * 2 + 1] = i1;
    aw[b * 2 + 0] = p[i0] / vs; aw[b * 2 + 1] = p[i1] / vs;
  }
}

// ======================================================================
// 3) attention core: one block per (sample, slot, head)
// ======================================================================
__global__ __launch_bounds__(256) void k_attn_core(
    const unsigned short* __restrict__ xbf,
    const unsigned short* __restrict__ WqkvB,    // bf16 [E][3D][D]
    const float* __restrict__ bqkv, const int* __restrict__ aidx,
    unsigned short* __restrict__ osb) {
  __shared__ unsigned short qh[Ss][DHh];   // Q, then reused for softmax probs
  __shared__ unsigned short kh[Ss][DHh];   // K
  __shared__ unsigned short vt[DHh][Ss];   // V^T (stored transposed from frags)
  __shared__ float sc[Ss][DHh];            // scores f32

  int blk  = blockIdx.x;
  int h    = blk & (Hh - 1);
  int pair = blk >> 3;
  int b    = pair >> 1;
  int e    = aidx[pair];
  int tid  = threadIdx.x;
  int wave = tid >> 5, lane = tid & 31, lo = lane & 15, hi = lane >> 4;

  const unsigned short* xb = xbf + (size_t)b * Ss * Dd;
  const unsigned short* We = WqkvB + (size_t)e * (3 * Dd) * Dd;

  // wave's two tiles: same M strip, adjacent N tiles
  int t0 = wave * 2;
  int mt = t0 >> 2, nt0 = t0 & 3, nt1 = nt0 + 1;

  // ---- Q/K/V projections fused over shared A: 6 accumulators ----
  {
    const unsigned short* Wq = We + (size_t)(0 * Dd + h * DHh) * Dd;
    const unsigned short* Wk = We + (size_t)(1 * Dd + h * DHh) * Dd;
    const unsigned short* Wv = We + (size_t)(2 * Dd + h * DHh) * Dd;
    v8f aq0 = zero8(), aq1 = zero8();
    v8f ak0 = zero8(), ak1 = zero8();
    v8f av0 = zero8(), av1 = zero8();
#pragma unroll 2
    for (int kk = 0; kk < Dd / 32; ++kk) {
      v16bf a = load_frag_bf(xb, Dd, mt * 16, kk * 32, lo, hi);
      __builtin_prefetch(Wq + (size_t)(nt0 * 16 + lo) * Dd + kk * 32 + 64, 0, 0);
      v16bf bm;
      bm = load_frag_bf(Wq, Dd, nt0 * 16, kk * 32, lo, hi); aq0 = wmma_bf(a, bm, aq0);
      bm = load_frag_bf(Wq, Dd, nt1 * 16, kk * 32, lo, hi); aq1 = wmma_bf(a, bm, aq1);
      bm = load_frag_bf(Wk, Dd, nt0 * 16, kk * 32, lo, hi); ak0 = wmma_bf(a, bm, ak0);
      bm = load_frag_bf(Wk, Dd, nt1 * 16, kk * 32, lo, hi); ak1 = wmma_bf(a, bm, ak1);
      bm = load_frag_bf(Wv, Dd, nt0 * 16, kk * 32, lo, hi); av0 = wmma_bf(a, bm, av0);
      bm = load_frag_bf(Wv, Dd, nt1 * 16, kk * 32, lo, hi); av1 = wmma_bf(a, bm, av1);
    }
    const float* bias = bqkv + (size_t)e * 3 * Dd + h * DHh;
    int col0 = nt0 * 16 + lo, col1 = nt1 * 16 + lo;
    float bq0 = bias[col0],          bq1 = bias[col1];
    float bk0 = bias[Dd + col0],     bk1 = bias[Dd + col1];
    float bv0 = bias[2 * Dd + col0], bv1 = bias[2 * Dd + col1];
#pragma unroll
    for (int v = 0; v < 8; ++v) {
      int row = mt * 16 + v + 8 * hi;
      qh[row][col0] = f2bf_u(aq0[v] + bq0);
      qh[row][col1] = f2bf_u(aq1[v] + bq1);
      kh[row][col0] = f2bf_u(ak0[v] + bk0);
      kh[row][col1] = f2bf_u(ak1[v] + bk1);
    }
    // V^T: per lane, 8 consecutive rows at fixed column -> one b128 store
    uint4 pk0, pk1;
    pk0.x = pack2(av0[0] + bv0, av0[1] + bv0);
    pk0.y = pack2(av0[2] + bv0, av0[3] + bv0);
    pk0.z = pack2(av0[4] + bv0, av0[5] + bv0);
    pk0.w = pack2(av0[6] + bv0, av0[7] + bv0);
    pk1.x = pack2(av1[0] + bv1, av1[1] + bv1);
    pk1.y = pack2(av1[2] + bv1, av1[3] + bv1);
    pk1.z = pack2(av1[4] + bv1, av1[5] + bv1);
    pk1.w = pack2(av1[6] + bv1, av1[7] + bv1);
    *(uint4*)&vt[col0][mt * 16 + 8 * hi] = pk0;
    *(uint4*)&vt[col1][mt * 16 + 8 * hi] = pk1;
  }
  __syncthreads();

  // ---- scores = Q K^T * dh^-0.5 ----
  {
    v8f acc0 = zero8(), acc1 = zero8();
#pragma unroll
    for (int kk = 0; kk < 2; ++kk) {
      v16bf a  = load_frag_bf(&qh[0][0], DHh, mt * 16, kk * 32, lo, hi);
      v16bf b0 = load_frag_bf(&kh[0][0], DHh, nt0 * 16, kk * 32, lo, hi);
      acc0 = wmma_bf(a, b0, acc0);
      v16bf b1 = load_frag_bf(&kh[0][0], DHh, nt1 * 16, kk * 32, lo, hi);
      acc1 = wmma_bf(a, b1, acc1);
    }
#pragma unroll
    for (int v = 0; v < 8; ++v) {
      int row = mt * 16 + v + 8 * hi;
      sc[row][nt0 * 16 + lo] = acc0[v] * 0.125f;
      sc[row][nt1 * 16 + lo] = acc1[v] * 0.125f;
    }
  }
  __syncthreads();

  // ---- softmax rows (probs overwrite Q buffer) ----
  if (tid < Ss) {
    float mx = -1e30f;
    for (int c = 0; c < Ss; ++c) mx = fmaxf(mx, sc[tid][c]);
    float sum = 0.f;
    for (int c = 0; c < Ss; ++c) { float v = expf(sc[tid][c] - mx); sc[tid][c] = v; sum += v; }
    float inv = 1.0f / sum;
    for (int c = 0; c < Ss; c += 2)
      *(unsigned*)&qh[tid][c] = pack2(sc[tid][c] * inv, sc[tid][c + 1] * inv);
  }
  __syncthreads();

  // ---- O = P V ----
  {
    v8f acc0 = zero8(), acc1 = zero8();
#pragma unroll
    for (int kk = 0; kk < 2; ++kk) {
      v16bf a  = load_frag_bf(&qh[0][0], DHh, mt * 16, kk * 32, lo, hi);  // probs
      v16bf b0 = load_frag_bf(&vt[0][0], Ss,  nt0 * 16, kk * 32, lo, hi); // V^T
      acc0 = wmma_bf(a, b0, acc0);
      v16bf b1 = load_frag_bf(&vt[0][0], Ss,  nt1 * 16, kk * 32, lo, hi);
      acc1 = wmma_bf(a, b1, acc1);
    }
    unsigned short* op = osb + ((size_t)pair * Ss) * Dd + h * DHh;
#pragma unroll
    for (int v = 0; v < 8; ++v) {
      size_t row = (size_t)(mt * 16 + v + 8 * hi) * Dd;
      op[row + nt0 * 16 + lo] = f2bf_u(acc0[v]);
      op[row + nt1 * 16 + lo] = f2bf_u(acc1[v]);
    }
  }
}

// ======================================================================
// 4) attention projections: P2 = (O@Wo.T+bo)@Wp.T+bp ; y1 += w*P2
// ======================================================================
__global__ __launch_bounds__(256) void k_attn_proj(
    const unsigned short* __restrict__ osb,
    const unsigned short* __restrict__ WoB, const float* __restrict__ bo,
    const unsigned short* __restrict__ WpB, const float* __restrict__ bp,
    const int* __restrict__ aidx, const float* __restrict__ aw,
    float* __restrict__ y1) {
  __shared__ unsigned short p1[Ss][Dd];    // 64KB bf16 intermediate
  int pair = blockIdx.x;
  int b = pair >> 1;
  int e = aidx[pair];
  float wgt = aw[pair];
  int tid = threadIdx.x;
  int wave = tid >> 5, lane = tid & 31, lo = lane & 15, hi = lane >> 4;

  const unsigned short* ob = osb + (size_t)pair * Ss * Dd;
  const unsigned short* WoE = WoB + (size_t)e * Dd * Dd;
  const float* boE = bo + (size_t)e * Dd;

  // P1 = O @ Wo.T + bo  -> bf16 LDS
  for (int mt = 0; mt < 4; ++mt) {
    v8f acc[4] = {zero8(), zero8(), zero8(), zero8()};
#pragma unroll 2
    for (int kk = 0; kk < Dd / 32; ++kk) {
      v16bf a = load_frag_bf(ob, Dd, mt * 16, kk * 32, lo, hi);
      __builtin_prefetch(WoE + (size_t)(wave * 64 + lo) * Dd + kk * 32 + 64, 0, 0);
#pragma unroll
      for (int nn = 0; nn < 4; ++nn) {
        v16bf bm = load_frag_bf(WoE, Dd, (wave * 4 + nn) * 16, kk * 32, lo, hi);
        acc[nn] = wmma_bf(a, bm, acc[nn]);
      }
    }
#pragma unroll
    for (int nn = 0; nn < 4; ++nn) {
      int n0 = (wave * 4 + nn) * 16 + lo;
      float bn = boE[n0];
#pragma unroll
      for (int v = 0; v < 8; ++v)
        p1[mt * 16 + v + 8 * hi][n0] = f2bf_u(acc[nn][v] + bn);
    }
  }
  __syncthreads();

  // P2 = P1 @ Wp.T + bp ; y1 += w*P2
  const unsigned short* WpE = WpB + (size_t)e * Dd * Dd;
  const float* bpE = bp + (size_t)e * Dd;
  float* yb = y1 + (size_t)b * Ss * Dd;
  for (int mt = 0; mt < 4; ++mt) {
    v8f acc[4] = {zero8(), zero8(), zero8(), zero8()};
#pragma unroll 2
    for (int kk = 0; kk < Dd / 32; ++kk) {
      v16bf a = load_frag_bf(&p1[0][0], Dd, mt * 16, kk * 32, lo, hi);
      __builtin_prefetch(WpE + (size_t)(wave * 64 + lo) * Dd + kk * 32 + 64, 0, 0);
#pragma unroll
      for (int nn = 0; nn < 4; ++nn) {
        v16bf bm = load_frag_bf(WpE, Dd, (wave * 4 + nn) * 16, kk * 32, lo, hi);
        acc[nn] = wmma_bf(a, bm, acc[nn]);
      }
    }
#pragma unroll
    for (int nn = 0; nn < 4; ++nn) {
      int n0 = (wave * 4 + nn) * 16 + lo;
      float bn = bpE[n0];
#pragma unroll
      for (int v = 0; v < 8; ++v)
        atomicAdd(&yb[(size_t)(mt * 16 + v + 8 * hi) * Dd + n0],
                  wgt * (acc[nn][v] + bn));
    }
  }
}

// ======================================================================
// 5/8) batchnorm stats: per-channel sum / sumsq (atomic partials)
// ======================================================================
__global__ __launch_bounds__(256) void k_bn_stats(
    const float* __restrict__ y, float* s1, float* s2) {
  int tid = threadIdx.x;
  int r0 = blockIdx.x * 32;
  float a0 = 0.f, a1 = 0.f, q0 = 0.f, q1 = 0.f;
  for (int r = 0; r < 32; ++r) {
    const float* p = y + (size_t)(r0 + r) * Dd;
    float v0 = p[tid], v1 = p[tid + 256];
    a0 += v0; q0 += v0 * v0;
    a1 += v1; q1 += v1 * v1;
  }
  atomicAdd(&s1[tid], a0);       atomicAdd(&s1[tid + 256], a1);
  atomicAdd(&s2[tid], q0);       atomicAdd(&s2[tid + 256], q1);
}

// ======================================================================
// 6) BN1 apply + FFN router + top2 + expert bucket scatter
// ======================================================================
__global__ __launch_bounds__(256) void k_bn1_router(
    const float* __restrict__ y1, const float* __restrict__ sumA,
    const float* __restrict__ sqA, const float* __restrict__ g1,
    const float* __restrict__ beta1, const float* __restrict__ Wrf,
    const float* __restrict__ brf, float* __restrict__ y2,
    unsigned short* __restrict__ zbf, float* out_pf,
    int* cnt, int* flist, float* fwv) {
  __shared__ float zs[16][Dd];    // 32KB
  __shared__ float lgs[16][EFx];
  int tid = threadIdx.x;
  int t0 = blockIdx.x * 16;
  const float inv_n = 1.0f / (float)TOKS;
#pragma unroll
  for (int dd = 0; dd < 2; ++dd) {
    int d = tid + dd * 256;
    float mu = sumA[d] * inv_n;
    float var = sqA[d] * inv_n - mu * mu;
    float rs = rsqrtf(var + EPSV) * g1[d];
    float bb = beta1[d];
    for (int tk = 0; tk < 16; ++tk) {
      size_t idx = (size_t)(t0 + tk) * Dd + d;
      float z = (y1[idx] - mu) * rs + bb;
      zs[tk][d] = z;
      y2[idx] = z;            // residual base for FFN accumulation
      zbf[idx] = f2bf_u(z);
    }
  }
  __syncthreads();
  if (tid < 128) {
    int tk = tid >> 3, e = tid & 7;
    const float* wr = Wrf + e * Dd;
    float s = brf[e];
    for (int d = 0; d < Dd; ++d) s += zs[tk][d] * wr[d];
    lgs[tk][e] = s;
  }
  __syncthreads();
  if (tid < 16) {
    int t = t0 + tid;
    int srow = t & (Ss - 1);
    float p[EFx];
    float mx = lgs[tid][0];
    for (int e = 1; e < EFx; ++e) mx = fmaxf(mx, lgs[tid][e]);
    float ss = 0.f;
    for (int e = 0; e < EFx; ++e) { p[e] = expf(lgs[tid][e] - mx); ss += p[e]; }
    float inv = 1.0f / ss;
    for (int e = 0; e < EFx; ++e) {
      p[e] *= inv;
      atomicAdd(&out_pf[srow * EFx + e], p[e] * (1.0f / Bb));
    }
    int i0 = 0;
    for (int e = 1; e < EFx; ++e) if (p[e] > p[i0]) i0 = e;
    int i1 = (i0 == 0) ? 1 : 0;
    for (int e = 0; e < EFx; ++e) if (e != i0 && p[e] > p[i1]) i1 = e;
    float vs = p[i0] + p[i1];
    int pos0 = atomicAdd(&cnt[i0], 1);
    flist[i0 * (2 * TOKS) + pos0] = t * 2 + 0;
    fwv[t * 2 + 0] = p[i0] / vs;
    int pos1 = atomicAdd(&cnt[i1], 1);
    flist[i1 * (2 * TOKS) + pos1] = t * 2 + 1;
    fwv[t * 2 + 1] = p[i1] / vs;
  }
}

// ======================================================================
// 7) FFN experts: gathered 64-token tiles staged to LDS once,
//    hidden streamed in 128-wide chunks through LDS
// ======================================================================
__global__ __launch_bounds__(256) void k_ffn(
    const unsigned short* __restrict__ zbf,
    const unsigned short* __restrict__ W1B, const float* __restrict__ b1,
    const unsigned short* __restrict__ W2B, const float* __restrict__ b2,
    const int* __restrict__ cnt, const int* __restrict__ flist,
    const float* __restrict__ fwv, float* __restrict__ y2) {
  __shared__ int   toks[64];
  __shared__ float wrs[64];
  __shared__ unsigned short zrow[64][Dd];  // 64KB gathered activations (bf16)
  __shared__ unsigned short hb[64][128];   // 16KB hidden chunk (post-GELU, bf16)

  int e = blockIdx.x >> 8;
  int tile = blockIdx.x & 255;
  int ne = cnt[e];
  int base = tile * 64;
  if (base >= ne) return;

  int tid = threadIdx.x;
  int wave = tid >> 5, lane = tid & 31, lo = lane & 15, hi = lane >> 4;

  if (tid < 64) {
    int idx = base + tid;
    if (idx < ne) {
      int en = flist[e * (2 * TOKS) + idx];
      toks[tid] = en >> 1;
      wrs[tid] = fwv[en];
    } else { toks[tid] = 0; wrs[tid] = 0.0f; }   // pad rows: weight 0
  }
  __syncthreads();

  // stage gathered rows into LDS once (reused by all 16 hidden chunks)
  {
    int r = tid >> 2;
    int seg = (tid & 3) * 128;
    const unsigned short* src = zbf + (size_t)toks[r] * Dd + seg;
    unsigned short* dst = &zrow[r][seg];
#pragma unroll
    for (int i = 0; i < 128; i += 8)
      *(uint4*)(dst + i) = *(const uint4*)(src + i);
  }
  __syncthreads();

  const unsigned short* W1e = W1B + (size_t)e * DFF * Dd;
  const float* b1e = b1 + (size_t)e * DFF;
  const unsigned short* W2e = W2B + (size_t)e * Dd * DFF;
  const float* b2e = b2 + (size_t)e * Dd;

  v8f oacc[4][4];
#pragma unroll
  for (int a = 0; a < 4; ++a)
#pragma unroll
    for (int c = 0; c < 4; ++c) oacc[a][c] = zero8();

  for (int hc = 0; hc < DFF / 128; ++hc) {
    // hidden chunk: wave owns 16-wide column strip (wave*16)
    v8f ha[4] = {zero8(), zero8(), zero8(), zero8()};
    int nrow0 = hc * 128 + wave * 16;
#pragma unroll 2
    for (int kk = 0; kk < Dd / 32; ++kk) {
      __builtin_prefetch(W1e + (size_t)(nrow0 + lo) * Dd + kk * 32 + 64, 0, 0);
      v16bf bm = load_frag_bf(W1e, Dd, nrow0, kk * 32, lo, hi);
#pragma unroll
      for (int mt = 0; mt < 4; ++mt) {
        v16bf a = load_frag_bf(&zrow[0][0], Dd, mt * 16, kk * 32, lo, hi);
        ha[mt] = wmma_bf(a, bm, ha[mt]);
      }
    }
    float bn = b1e[nrow0 + lo];
#pragma unroll
    for (int mt = 0; mt < 4; ++mt)
#pragma unroll
      for (int v = 0; v < 8; ++v) {
        float xv = ha[mt][v] + bn;
        float g = 0.5f * xv * (1.0f + erff(xv * 0.70710678118f));  // exact GELU
        hb[mt * 16 + v + 8 * hi][wave * 16 + lo] = f2bf_u(g);
      }
    __syncthreads();
    // accumulate out += H_chunk @ W2_chunk.T
#pragma unroll
    for (int kk = 0; kk < 4; ++kk) {
      v16bf af[4];
#pragma unroll
      for (int mt = 0; mt < 4; ++mt)
        af[mt] = load_frag_bf(&hb[0][0], 128, mt * 16, kk * 32, lo, hi);
#pragma unroll
      for (int nn = 0; nn < 4; ++nn) {
        v16bf bm = load_frag_bf(W2e, DFF, (wave * 4 + nn) * 16,
                                hc * 128 + kk * 32, lo, hi);
#pragma unroll
        for (int mt = 0; mt < 4; ++mt)
          oacc[mt][nn] = wmma_bf(af[mt], bm, oacc[mt][nn]);
      }
    }
    __syncthreads();
  }

  // finalize: weighted scatter-accumulate into y2
#pragma unroll
  for (int mt = 0; mt < 4; ++mt)
#pragma unroll
    for (int nn = 0; nn < 4; ++nn) {
      int n0 = (wave * 4 + nn) * 16 + lo;
      float bn = b2e[n0];
#pragma unroll
      for (int v = 0; v < 8; ++v) {
        int m = mt * 16 + v + 8 * hi;
        float r = wrs[m] * (oacc[mt][nn][v] + bn);
        atomicAdd(&y2[(size_t)toks[m] * Dd + n0], r);
      }
    }
}

// ======================================================================
// 9) BN2 apply -> final output
// ======================================================================
__global__ __launch_bounds__(256) void k_bn2_apply(
    const float* __restrict__ y2, const float* __restrict__ sumF,
    const float* __restrict__ sqF, const float* __restrict__ g2,
    const float* __restrict__ beta2, float* __restrict__ out) {
  size_t stride = (size_t)gridDim.x * blockDim.x;
  const float inv_n = 1.0f / (float)TOKS;
  for (size_t i = (size_t)blockIdx.x * blockDim.x + threadIdx.x;
       i < (size_t)TOKS * Dd; i += stride) {
    int d = (int)(i & (Dd - 1));
    float mu = sumF[d] * inv_n;
    float var = sqF[d] * inv_n - mu * mu;
    out[i] = (y2[i] - mu) * rsqrtf(var + EPSV) * g2[d] + beta2[d];
  }
}

// ======================================================================
extern "C" void kernel_launch(void* const* d_in, const int* in_sizes, int n_in,
                              void* d_out, int out_size, void* d_ws, size_t ws_size,
                              hipStream_t stream) {
  (void)in_sizes; (void)n_in; (void)out_size; (void)ws_size;
  const float* x    = (const float*)d_in[0];
  const float* Wqkv = (const float*)d_in[1];
  const float* bqkv = (const float*)d_in[2];
  const float* Wo   = (const float*)d_in[3];
  const float* bo   = (const float*)d_in[4];
  const float* Wp   = (const float*)d_in[5];
  const float* bp   = (const float*)d_in[6];
  const float* W1   = (const float*)d_in[7];
  const float* b1   = (const float*)d_in[8];
  const float* W2   = (const float*)d_in[9];
  const float* b2   = (const float*)d_in[10];
  const float* Wra  = (const float*)d_in[11];
  const float* bra  = (const float*)d_in[12];
  const float* Wrf  = (const float*)d_in[13];
  const float* brf  = (const float*)d_in[14];
  const float* g1   = (const float*)d_in[15];
  const float* beta1= (const float*)d_in[16];
  const float* g2   = (const float*)d_in[17];
  const float* beta2= (const float*)d_in[18];

  float* out    = (float*)d_out;
  float* out_pa = out + (size_t)TOKS * Dd;   // [8]
  float* out_pf = out_pa + EAx;              // [64,8]

  // workspace carve-out (256B aligned)
  char* w = (char*)d_ws;
  auto carve = [&](size_t bytes) -> void* {
    void* p = (void*)w;
    w += (bytes + 255) & ~(size_t)255;
    return p;
  };
  const size_t nWqkv = (size_t)EAx * 3 * Dd * Dd;   // 6.3M
  const size_t nWo   = (size_t)EAx * Dd * Dd;       // 2.1M
  const size_t nW1   = (size_t)EFx * DFF * Dd;      // 8.4M
  unsigned short* xbf  = (unsigned short*)carve((size_t)TOKS * Dd * 2);       // 8MB
  unsigned short* zbf  = (unsigned short*)carve((size_t)TOKS * Dd * 2);       // 8MB
  unsigned short* osb  = (unsigned short*)carve((size_t)Bb * 2 * Ss * Dd * 2);// 16MB
  float* y1   = (float*)carve((size_t)TOKS * Dd * 4);                         // 16MB
  float* y2   = (float*)carve((size_t)TOKS * Dd * 4);                         // 16MB
  unsigned short* wqkvB = (unsigned short*)carve(nWqkv * 2);                  // 12.6MB
  unsigned short* woB   = (unsigned short*)carve(nWo * 2);                    // 4.2MB
  unsigned short* wpB   = (unsigned short*)carve(nWo * 2);                    // 4.2MB
  unsigned short* w1B   = (unsigned short*)carve(nW1 * 2);                    // 16.8MB
  unsigned short* w2B   = (unsigned short*)carve(nW1 * 2);                    // 16.8MB
  int*   aidx = (int*)carve(Bb * 2 * 4);
  float* aw   = (float*)carve(Bb * 2 * 4);
  float* sumA = (float*)carve(Dd * 4);
  float* sqA  = (float*)carve(Dd * 4);
  float* sumF = (float*)carve(Dd * 4);
  float* sqF  = (float*)carve(Dd * 4);
  int*   cnt  = (int*)carve(EFx * 4);
  int*   flist= (int*)carve((size_t)EFx * 2 * TOKS * 4);                      // 512KB
  float* fwv  = (float*)carve((size_t)2 * TOKS * 4);                          // 64KB

  // one-time (per launch) weight downconversion fp32 -> bf16
  k_cvt<<<2048, 256, 0, stream>>>(Wqkv, (unsigned*)wqkvB, nWqkv / 2);
  k_cvt<<<1024, 256, 0, stream>>>(Wo,   (unsigned*)woB,   nWo / 2);
  k_cvt<<<1024, 256, 0, stream>>>(Wp,   (unsigned*)wpB,   nWo / 2);
  k_cvt<<<2048, 256, 0, stream>>>(W1,   (unsigned*)w1B,   nW1 / 2);
  k_cvt<<<2048, 256, 0, stream>>>(W2,   (unsigned*)w2B,   nW1 / 2);

  k_prep<<<4096, 256, 0, stream>>>(x, xbf, y1, out_pa, out_pf,
                                   sumA, sqA, sumF, sqF, cnt);
  k_router_attn<<<Bb, 256, 0, stream>>>(x, Wra, bra, aidx, aw, out_pa);
  k_attn_core<<<Bb * 2 * Hh, 256, 0, stream>>>(xbf, wqkvB, bqkv, aidx, osb);
  k_attn_proj<<<Bb * 2, 256, 0, stream>>>(osb, woB, bo, wpB, bp, aidx, aw, y1);
  k_bn_stats<<<TOKS / 32, 256, 0, stream>>>(y1, sumA, sqA);
  k_bn1_router<<<TOKS / 16, 256, 0, stream>>>(y1, sumA, sqA, g1, beta1,
                                              Wrf, brf, y2, zbf, out_pf,
                                              cnt, flist, fwv);
  k_ffn<<<EFx * 256, 256, 0, stream>>>(zbf, w1B, b1, w2B, b2,
                                       cnt, flist, fwv, y2);
  k_bn_stats<<<TOKS / 32, 256, 0, stream>>>(y2, sumF, sqF);
  k_bn2_apply<<<4096, 256, 0, stream>>>(y2, sumF, sqF, g2, beta2, out);
}
